// ETAL_56873956934472
// MI455X (gfx1250) — matrix operand order repaired and checked by reading, not compiled, not run on hardware
//
#include <hip/hip_runtime.h>

#define BB 4
#define NNH 8
#define LL 1024
#define DD 256
#define EE 112
#define BN (BB*NNH)

typedef __attribute__((ext_vector_type(16))) __bf16 v16bf;
typedef __attribute__((ext_vector_type(8)))  __bf16 v8bf;
typedef __attribute__((ext_vector_type(8)))  float  v8f;

// ---------------------------------------------------------------- PE helper
__device__ __forceinline__ float pe_val(int j, int i, float inv_dim) {
    float expo = (float)(i & ~1) * inv_dim;
    float ang  = (float)j * __powf(10000.0f, -expo);
    return (i & 1) ? __cosf(ang) : __sinf(ang);
}

// ---------------------------------------------------------------- WMMA helpers
__device__ __forceinline__ v8f wmma_bf16(v16bf a, v16bf b, v8f c) {
    return __builtin_amdgcn_wmma_f32_16x16x32_bf16(
        false, a, false, b, (short)0, c, false, false);
}

// A-matrix 16x32 bf16 fragment (row-major source, ld elements per row).
// lane<16: row = row0+lane, K = {0..7, 16..23}; lane>=16: same row, K = {8..15, 24..31}
__device__ __forceinline__ v16bf load_afrag(const __bf16* base, int ld, int row0, int k0) {
    int lane = threadIdx.x & 31;
    int half = lane >> 4, r = lane & 15;
    const __bf16* p = base + (size_t)(row0 + r) * ld + k0 + half * 8;
    v8bf lo = *(const v8bf*)p;
    v8bf hi = *(const v8bf*)(p + 16);
    v16bf a;
#pragma unroll
    for (int i = 0; i < 8; ++i) { a[i] = lo[i]; a[8 + i] = hi[i]; }
    return a;
}

// B-matrix 32x16 bf16 fragment. Source row-major with "columns of B" = rows of src:
// B[k][n] = src[col0+n][k0+k].  lane<16: n=lane, K={0..7,16..23}; lane>=16: n=lane-16, K={8..15,24..31}
__device__ __forceinline__ v16bf load_bfrag(const __bf16* base, int ld, int col0, int k0) {
    int lane = threadIdx.x & 31;
    int half = lane >> 4, c = lane & 15;
    const __bf16* p = base + (size_t)(col0 + c) * ld + k0 + half * 8;
    v8bf lo = *(const v8bf*)p;
    v8bf hi = *(const v8bf*)(p + 16);
    v16bf b;
#pragma unroll
    for (int i = 0; i < 8; ++i) { b[i] = lo[i]; b[8 + i] = hi[i]; }
    return b;
}

// ---------------------------------------------------------------- prep kernels
__global__ void k_theta_pe(const float* __restrict__ theta, __bf16* __restrict__ out) {
    size_t idx = (size_t)blockIdx.x * 256 + threadIdx.x;   // B*N*L*D threads exactly
    int dk = (int)(idx & 255u);
    int l  = (int)((idx >> 8) & 1023u);
    out[idx] = (__bf16)(theta[idx] + pe_val(l, dk, 1.0f / 256.0f));
}

__global__ void k_w_conv(const float* __restrict__ Wq, const float* __restrict__ Wk,
                         const float* __restrict__ Wv, __bf16* __restrict__ out) {
    int idx = blockIdx.x * 256 + threadIdx.x;              // 3*256*256 threads
    int w = idx >> 16;
    int r = idx & 65535;
    const float* src = (w == 0) ? Wq : (w == 1) ? Wk : Wv;
    out[idx] = (__bf16)src[r];
}

__global__ void k_edge(const float* __restrict__ e, float* __restrict__ out) {
    int idx = blockIdx.x * 256 + threadIdx.x;              // B*L*112 threads
    int c = idx % EE;
    int l = (idx / EE) % LL;
    out[idx] = e[idx] + pe_val(l, c, 1.0f / 112.0f);
}

// ---------------------------------------------------------------- projection GEMM
// out[l][e] = sum_d thetaPE[l][d] * W[e][d]  (bf16 in, f32 WMMA acc, bf16 out)
// Q,K stored [bn][l][e]; V stored transposed [bn][e][l] for contiguous PV B-frags.
__global__ void k_project(const __bf16* __restrict__ thetaPE, const __bf16* __restrict__ Wbf,
                          __bf16* __restrict__ Q, __bf16* __restrict__ K,
                          __bf16* __restrict__ Vt) {
    int wave = (int)((blockIdx.x * blockDim.x + threadIdx.x) >> 5);
    int lane = threadIdx.x & 31;
    int et = wave & 15;
    int lt = (wave >> 4) & 63;
    int bn = (wave >> 10) & 31;
    int w  = wave >> 15;                                   // 0=Q 1=K 2=V

    const __bf16* A  = thetaPE + (size_t)bn * LL * DD;
    const __bf16* Wm = Wbf + (size_t)w * DD * DD;

    v8f acc = {};
#pragma unroll
    for (int dc = 0; dc < 8; ++dc) {
        v16bf a = load_afrag(A, DD, lt * 16, dc * 32);
        v16bf b = load_bfrag(Wm, DD, et * 16, dc * 32);
        acc = wmma_bf16(a, b, acc);
    }
    int half = lane >> 4, c = lane & 15;
    int e = et * 16 + c;
#pragma unroll
    for (int r = 0; r < 8; ++r) {
        int l = lt * 16 + r + half * 8;
        __bf16 v = (__bf16)acc[r];
        if (w == 0)      Q[((size_t)bn * LL + l) * DD + e] = v;
        else if (w == 1) K[((size_t)bn * LL + l) * DD + e] = v;
        else             Vt[((size_t)bn * DD + e) * LL + l] = v;
    }
}

// ---------------------------------------------------------------- flash attention
#define ATT_WAVES 2
__global__ void k_attn(const __bf16* __restrict__ Q, const __bf16* __restrict__ K,
                       const __bf16* __restrict__ Vt, float* __restrict__ h_out) {
    __shared__ __bf16 ldsP[ATT_WAVES][16 * 32];
    int warp = threadIdx.x >> 5;
    int lane = threadIdx.x & 31;
    int tile = blockIdx.x * ATT_WAVES + warp;              // B*N*(L/16) = 2048 tiles
    int qt = tile & 63;
    int bn = tile >> 6;
    int q0 = qt * 16;
    const __bf16* Qb = Q  + (size_t)bn * LL * DD;
    const __bf16* Kb = K  + (size_t)bn * LL * DD;
    const __bf16* Vb = Vt + (size_t)bn * DD * LL;
    __bf16* lp = &ldsP[warp][0];
    int half = lane >> 4, c = lane & 15;

    // resident Q fragments: 16 rows x 256 d
    v16bf qf[8];
#pragma unroll
    for (int dc = 0; dc < 8; ++dc) qf[dc] = load_afrag(Qb, DD, q0, dc * 32);

    v8f zero = {};
    v8f acc[16];
#pragma unroll
    for (int nt = 0; nt < 16; ++nt) acc[nt] = zero;
    float rmax[8], rsum[8];
#pragma unroll
    for (int r = 0; r < 8; ++r) { rmax[r] = -__builtin_inff(); rsum[r] = 0.0f; }

    int kb_max = (q0 + 15) >> 5;
    for (int kb = 0; kb <= kb_max; ++kb) {
        int k0 = kb * 32;
        // S = Q K^T over d=256  (two 16x16 tiles -> 32 key columns)
        v8f s0 = zero, s1 = zero;
#pragma unroll
        for (int dc = 0; dc < 8; ++dc) {
            s0 = wmma_bf16(qf[dc], load_bfrag(Kb, DD, k0,      dc * 32), s0);
            s1 = wmma_bf16(qf[dc], load_bfrag(Kb, DD, k0 + 16, dc * 32), s1);
        }
        const float sc = 0.0625f;                          // 1/sqrt(256)
        bool need_mask = (k0 + 31 > q0);
#pragma unroll
        for (int r = 0; r < 8; ++r) {
            float a = s0[r] * sc, b = s1[r] * sc;
            int row = q0 + r + half * 8;
            if (need_mask) {
                if (k0 + c > row)      a = -__builtin_inff();
                if (k0 + 16 + c > row) b = -__builtin_inff();
            }
            s0[r] = a; s1[r] = b;
        }
        // online softmax; each row lives in one 16-lane half -> xor-reduce over 4 bits
        float p0[8], p1[8];
#pragma unroll
        for (int r = 0; r < 8; ++r) {
            float m = fmaxf(s0[r], s1[r]);
#pragma unroll
            for (int off = 1; off < 16; off <<= 1) m = fmaxf(m, __shfl_xor(m, off, 32));
            float nm   = fmaxf(rmax[r], m);
            float corr = __expf(rmax[r] - nm);
            rmax[r] = nm;
            float e0 = __expf(s0[r] - nm);
            float e1 = __expf(s1[r] - nm);
            p0[r] = e0; p1[r] = e1;
            float ps = e0 + e1;
#pragma unroll
            for (int off = 1; off < 16; off <<= 1) ps += __shfl_xor(ps, off, 32);
            rsum[r] = rsum[r] * corr + ps;
#pragma unroll
            for (int nt = 0; nt < 16; ++nt) acc[nt][r] *= corr;
        }
        // D-layout P -> A-layout via LDS transpose (per-wave buffer, in-order DS)
#pragma unroll
        for (int r = 0; r < 8; ++r) {
            int m = r + half * 8;
            lp[m * 32 + c]      = (__bf16)p0[r];
            lp[m * 32 + 16 + c] = (__bf16)p1[r];
        }
        asm volatile("s_wait_dscnt 0" ::: "memory");
        v16bf pf;
        {
            int r = lane & 15;
            const __bf16* pp = lp + r * 32 + half * 8;
            v8bf lo = *(const v8bf*)pp;
            v8bf hi = *(const v8bf*)(pp + 16);
#pragma unroll
            for (int i = 0; i < 8; ++i) { pf[i] = lo[i]; pf[8 + i] = hi[i]; }
        }
        // PV: V stored transposed -> contiguous B-fragments over the key dim
#pragma unroll
        for (int nt = 0; nt < 16; ++nt) {
            v16bf vf = load_bfrag(Vb, LL, nt * 16, k0);
            acc[nt] = wmma_bf16(pf, vf, acc[nt]);
        }
    }
    // epilogue: normalize + store f32
    float inv[8];
#pragma unroll
    for (int r = 0; r < 8; ++r) inv[r] = 1.0f / rsum[r];
    float* H = h_out + (size_t)bn * LL * DD;
#pragma unroll
    for (int nt = 0; nt < 16; ++nt)
#pragma unroll
        for (int r = 0; r < 8; ++r) {
            int row = q0 + r + half * 8;
            H[(size_t)row * DD + nt * 16 + c] = acc[nt][r] * inv[r];
        }
}

// ---------------------------------------------------------------- equivariant path
__global__ void k_equiv(const float* __restrict__ xi, float* __restrict__ x_out) {
    int wave = (int)((blockIdx.x * blockDim.x + threadIdx.x) >> 5);
    int lane = threadIdx.x & 31;
    int q  = wave & 1023;
    int bn = wave >> 10;
    const float* X = xi + (size_t)bn * LL * 3;
    const float inv23 = 0.0021544347f;                     // 10000^(-2/3)
    float xq0 = X[q * 3 + 0] + __sinf((float)q);
    float xq1 = X[q * 3 + 1] + __cosf((float)q);
    float xq2 = X[q * 3 + 2] + __sinf((float)q * inv23);

    float mloc = -__builtin_inff(), sloc = 0.f, a0 = 0.f, a1 = 0.f, a2 = 0.f;
    for (int k = lane; k <= q; k += 32) {
        float fk = (float)k;
        float y0 = X[k * 3 + 0] + __sinf(fk);
        float y1 = X[k * 3 + 1] + __cosf(fk);
        float y2 = X[k * 3 + 2] + __sinf(fk * inv23);
        float d0 = xq0 - y0, d1 = xq1 - y1, d2 = xq2 - y2;
        float sq = d0 * d0 + d1 * d1 + d2 * d2;
        if (sq > mloc) {
            float cc = __expf(mloc - sq);
            sloc *= cc; a0 *= cc; a1 *= cc; a2 *= cc;
            mloc = sq;
        }
        float w = __expf(sq - mloc);
        sloc += w;                                         // diagonal counts in normalizer
        if (k != q) { a0 += w * y0; a1 += w * y1; a2 += w * y2; }  // mbeta zeros diagonal
    }
    float M = mloc;
#pragma unroll
    for (int off = 1; off < 32; off <<= 1) M = fmaxf(M, __shfl_xor(M, off, 32));
    float cc = __expf(mloc - M);
    sloc *= cc; a0 *= cc; a1 *= cc; a2 *= cc;
#pragma unroll
    for (int off = 1; off < 32; off <<= 1) {
        sloc += __shfl_xor(sloc, off, 32);
        a0   += __shfl_xor(a0,   off, 32);
        a1   += __shfl_xor(a1,   off, 32);
        a2   += __shfl_xor(a2,   off, 32);
    }
    if (lane == 0) {
        float invS = 1.0f / sloc;
        float beta_diag = __expf(-M) * invS;               // sq(q,q)=0
        float smb = 1.0f - beta_diag;                      // sum of mbeta row
        float* O = x_out + ((size_t)bn * LL + q) * 3;
        O[0] = xq0 + 0.5f * a0 * invS - 0.5f * smb * xq0;
        O[1] = xq1 + 0.5f * a1 * invS - 0.5f * smb * xq1;
        O[2] = xq2 + 0.5f * a2 * invS - 0.5f * smb * xq2;
    }
}

// ---------------------------------------------------------------- launcher
extern "C" void kernel_launch(void* const* d_in, const int* in_sizes, int n_in,
                              void* d_out, int out_size, void* d_ws, size_t ws_size,
                              hipStream_t stream) {
    const float* theta = (const float*)d_in[0];
    const float* xi    = (const float*)d_in[1];
    const float* edge  = (const float*)d_in[2];
    const float* Wq    = (const float*)d_in[3];
    const float* Wk    = (const float*)d_in[4];
    const float* Wv    = (const float*)d_in[5];

    float* h_out = (float*)d_out;                          // [B,N,L,D]
    float* x_out = h_out + (size_t)BB * NNH * LL * DD;     // [B,N,L,3]
    float* e_out = x_out + (size_t)BB * NNH * LL * 3;      // [B,L,112]

    char* ws = (char*)d_ws;
    const size_t SZ = (size_t)BN * LL * DD * 2;            // one bf16 [bn,L,D] buffer
    __bf16* thetaPE = (__bf16*)(ws);
    __bf16* Qb      = (__bf16*)(ws + SZ);
    __bf16* Kb      = (__bf16*)(ws + 2 * SZ);
    __bf16* Vt      = (__bf16*)(ws + 3 * SZ);
    __bf16* Wbf     = (__bf16*)(ws + 4 * SZ);              // 3*256*256 bf16

    k_theta_pe<<<32768, 256, 0, stream>>>(theta, thetaPE);
    k_w_conv  <<<768,   256, 0, stream>>>(Wq, Wk, Wv, Wbf);
    k_edge    <<<1792,  256, 0, stream>>>(edge, e_out);
    k_equiv   <<<4096,  256, 0, stream>>>(xi, x_out);
    k_project <<<12288, 256, 0, stream>>>(thetaPE, Wbf, Qb, Kb, Vt);
    k_attn    <<<1024, 32 * ATT_WAVES, 0, stream>>>(Qb, Kb, Vt, h_out);
}
